// FasterRCNN_84670985273808
// MI455X (gfx1250) — matrix operand order repaired and checked by reading, not compiled, not run on hardware
//
#include <hip/hip_runtime.h>

typedef _Float16 f16;
typedef __attribute__((ext_vector_type(16))) _Float16 v16h;
typedef __attribute__((ext_vector_type(8)))  float    v8f;

#define Hdim 64
#define Wdim 64
#define Cdim 256
#define NPIX 4096
#define KCONV 2304
#define KT_CONV 72
#define NT_CONV 256        /* 4096/16 pixel tiles */
#define MT_CONV 16         /* 256/16 channel tiles */
#define NSCORE 36864
#define SORTN 65536
#define KPRE 2000
#define NMSW 63            /* ceil(2000/32) */
#define KPOST 256
#define NCLS 21
#define KHEAD 50176        /* 256*14*14 */
#define KT_HEAD 1568
#define MT_HEAD 7          /* 112/16, 105 padded */
#define NT_HEAD 16         /* 256/16 */
#define KSLICE 8           /* split-K slices for head GEMM */
#define KT_SLICE 196       /* 1568/8 */
#define MT_RPN 4           /* 64/16, 54 padded */
#define KT_RPN 8           /* 256/32 */

// ---------- fragment index helpers (CDNA5 wave32 WMMA f16 layouts) ----------
// A (16x32): lane L holds M=L%16; k = 16*(L/16) + h
__device__ __forceinline__ int a_frag_k(int lane, int h) {
    return ((lane >> 4) << 4) + h;
}
// B (32x16): lane L holds N=L%16; k = 16*(L/16) + h
__device__ __forceinline__ int b_frag_k(int lane, int h) {
    return ((lane >> 4) << 4) + h;
}

// ------------------------------ pack kernels --------------------------------
__global__ __launch_bounds__(256) void pack_conv_w(const float* __restrict__ w,
                                                   f16* __restrict__ ap) {
    int gid  = blockIdx.x * 256 + threadIdx.x;           // 16*72*512 threads
    int frag = gid >> 9, r = gid & 511;
    int lane = r >> 4, h = r & 15;
    int mt = frag / KT_CONV, kt = frag % KT_CONV;
    int m = mt * 16 + (lane & 15);
    int k = kt * 32 + a_frag_k(lane, h);
    ap[gid] = (f16)w[(size_t)m * KCONV + k];
}

__global__ __launch_bounds__(256) void pack_conv_x(const float* __restrict__ x,
                                                   f16* __restrict__ bp) {
    int gid  = blockIdx.x * 256 + threadIdx.x;           // 72*256*512 threads
    int frag = gid >> 9, r = gid & 511;
    int lane = r >> 4, h = r & 15;
    int kt = frag / NT_CONV, nt = frag % NT_CONV;
    int k = kt * 32 + b_frag_k(lane, h);
    int c = k / 9, rem = k % 9, ky = rem / 3, kx = rem % 3;
    int n = nt * 16 + (lane & 15);
    int iy = (n >> 6) + ky - 1, ix = (n & 63) + kx - 1;
    float v = 0.0f;
    if (iy >= 0 && iy < Hdim && ix >= 0 && ix < Wdim)
        v = x[(size_t)c * NPIX + iy * Wdim + ix];
    bp[gid] = (f16)v;
}

__global__ __launch_bounds__(256) void pack_rpn_w(const float* __restrict__ cls_w,
                                                  const float* __restrict__ tfm_w,
                                                  f16* __restrict__ ap) {
    int gid  = blockIdx.x * 256 + threadIdx.x;           // 4*8*512 threads
    int frag = gid >> 9, r = gid & 511;
    int lane = r >> 4, h = r & 15;
    int mt = frag / KT_RPN, kt = frag % KT_RPN;
    int m = mt * 16 + (lane & 15);                       // 0..63, rows 54..63 pad
    int k = kt * 32 + a_frag_k(lane, h);
    float v = 0.0f;
    if (m < 18)      v = cls_w[(size_t)m * Cdim + k];
    else if (m < 54) v = tfm_w[(size_t)(m - 18) * Cdim + k];
    ap[gid] = (f16)v;
}

__global__ __launch_bounds__(256) void pack_head_w(const float* __restrict__ w,
                                                   f16* __restrict__ ap) {
    int gid  = blockIdx.x * 256 + threadIdx.x;           // 7*1568*512 threads
    int frag = gid >> 9, r = gid & 511;
    int lane = r >> 4, h = r & 15;
    int mt = frag / KT_HEAD, kt = frag % KT_HEAD;
    int m = mt * 16 + (lane & 15);
    int k = kt * 32 + a_frag_k(lane, h);
    float v = (m < 105) ? w[(size_t)m * KHEAD + k] : 0.0f;
    ap[gid] = (f16)v;
}

// ------- conv implicit GEMM, register-blocked 16x64, f16 fragment output ----
__global__ __launch_bounds__(32) void conv_wmma(const f16* __restrict__ apack,
                                                const f16* __restrict__ bpack,
                                                const float* __restrict__ bias,
                                                f16* __restrict__ hpackB) {
    int mt = blockIdx.x, ntb = blockIdx.y, lane = threadIdx.x;   // grid 16 x 64
    const v16h* Af = (const v16h*)apack + (size_t)mt * KT_CONV * 32 + lane;
    const v16h* Bf = (const v16h*)bpack + (size_t)ntb * 4 * 32 + lane;
    v8f acc[4] = {};
#pragma unroll 2
    for (int kt = 0; kt < KT_CONV; ++kt) {
        int kpf = (kt + 2 < KT_CONV) ? kt + 2 : KT_CONV - 1;
        __builtin_prefetch((const void*)(Af + (size_t)kpf * 32), 0, 1);
        __builtin_prefetch((const void*)(Bf + (size_t)kpf * NT_CONV * 32), 0, 1);
        v16h a = Af[(size_t)kt * 32];
#pragma unroll
        for (int b = 0; b < 4; ++b) {
            v16h bb = Bf[(size_t)kt * NT_CONV * 32 + b * 32];
            acc[b] = __builtin_amdgcn_wmma_f32_16x16x32_f16(false, a, false, bb,
                                                            (short)0, acc[b], false, false);
        }
    }
    // write h (+bias) straight into f16 B-fragments for the RPN GEMM
    int hi = lane >> 4, nl = lane & 15, mloc = hi * 8;
    int kt2 = mt >> 1, lsub = ((mt & 1) << 4) + nl;
#pragma unroll
    for (int b = 0; b < 4; ++b) {
        int nt = ntb * 4 + b;
        f16* dst = hpackB + ((((size_t)kt2 * NT_CONV + nt) * 32 + lsub) << 4);
#pragma unroll
        for (int r = 0; r < 8; ++r) {
            int m = mt * 16 + mloc + r;
            dst[mloc + r] = (f16)(acc[b][r] + bias[m]);
        }
    }
}

// ---------------- RPN projection GEMM (64x4096, K=256) via WMMA -------------
__global__ __launch_bounds__(32) void rpn_wmma(const f16* __restrict__ apack,
                                               const f16* __restrict__ bpack,
                                               const float* __restrict__ cls_b,
                                               const float* __restrict__ tfm_b,
                                               float* __restrict__ out_cls,
                                               float* __restrict__ out_tfm,
                                               float* __restrict__ o54map) {
    int mt = blockIdx.x, ntb = blockIdx.y, lane = threadIdx.x;   // grid 4 x 64
    const v16h* Af = (const v16h*)apack + (size_t)mt * KT_RPN * 32 + lane;
    const v16h* Bf = (const v16h*)bpack + (size_t)ntb * 4 * 32 + lane;
    v8f acc[4] = {};
#pragma unroll 2
    for (int kt = 0; kt < KT_RPN; ++kt) {
        v16h a = Af[(size_t)kt * 32];
#pragma unroll
        for (int b = 0; b < 4; ++b) {
            v16h bb = Bf[(size_t)kt * NT_CONV * 32 + b * 32];
            acc[b] = __builtin_amdgcn_wmma_f32_16x16x32_f16(false, a, false, bb,
                                                            (short)0, acc[b], false, false);
        }
    }
    int hi = lane >> 4, nl = lane & 15, mloc = hi * 8;
#pragma unroll
    for (int b = 0; b < 4; ++b) {
        int p0 = (ntb * 4 + b) * 16 + nl;
#pragma unroll
        for (int r = 0; r < 8; ++r) {
            int o = mt * 16 + mloc + r;
            float v = acc[b][r];
            if (o < 18)      { v += cls_b[o];      out_cls[(size_t)o * NPIX + p0] = v; }
            else if (o < 54) { v += tfm_b[o - 18]; out_tfm[(size_t)(o - 18) * NPIX + p0] = v; }
            o54map[(size_t)o * NPIX + p0] = v;
        }
    }
}

// -------------------- sigmoid + anchor box decode ---------------------------
__global__ __launch_bounds__(256) void rpn_decode(const float* __restrict__ o54map,
                                                  const float* __restrict__ anchors,
                                                  float* __restrict__ scores,
                                                  float* __restrict__ bxA, float* __restrict__ byA,
                                                  float* __restrict__ bwA, float* __restrict__ bhA) {
    int gid = blockIdx.x * 256 + threadIdx.x;            // 36864 threads
    int a = gid >> 12, p = gid & 4095;
    float c0 = o54map[(size_t)(2 * a) * NPIX + p];
    float c1 = o54map[(size_t)(2 * a + 1) * NPIX + p];
    float obj = 1.0f / (1.0f + __expf(-(c0 - c1)));
    float t0 = o54map[(size_t)(18 + 4 * a + 0) * NPIX + p];
    float t1 = o54map[(size_t)(18 + 4 * a + 1) * NPIX + p];
    float t2 = o54map[(size_t)(18 + 4 * a + 2) * NPIX + p];
    float t3 = o54map[(size_t)(18 + 4 * a + 3) * NPIX + p];
    float ah = anchors[2 * a + 0], aw = anchors[2 * a + 1];
    float jj = (float)(p & 63), ii = (float)(p >> 6);
    float bx = jj + aw * (t1 - 0.5f) * (1.0f / 16.0f);
    float by = ii + ah * (t0 - 0.5f) * (1.0f / 16.0f);
    bx = fminf(fmaxf(bx, 0.0f), 63.0f);
    by = fminf(fmaxf(by, 0.0f), 63.0f);
    float bw = fminf(fmaxf(bx + aw * (1.0f / 16.0f) * __expf(t3), 0.0f), 63.0f) - bx;
    float bh = fminf(fmaxf(by + ah * (1.0f / 16.0f) * __expf(t2), 0.0f), 63.0f) - by;
    scores[gid] = obj; bxA[gid] = bx; byA[gid] = by; bwA[gid] = bw; bhA[gid] = bh;
}

// ----------------- single-block bitonic top-K (256 KB LDS) ------------------
__global__ __launch_bounds__(1024) void bitonic_topk(const float* __restrict__ scores,
                                                     const float* __restrict__ bxA, const float* __restrict__ byA,
                                                     const float* __restrict__ bwA, const float* __restrict__ bhA,
                                                     float* __restrict__ tbx, float* __restrict__ tby,
                                                     float* __restrict__ tbw, float* __restrict__ tbh) {
    extern __shared__ unsigned km[];
    int tid = threadIdx.x;
    for (int i = tid; i < SORTN; i += 1024) {
        unsigned key = 0xFFFFFFFFu;
        if (i < NSCORE) {
            unsigned b = __float_as_uint(scores[i]);      // sigmoid => positive, bits monotonic
            key = ((0xFFFFu - (b >> 16)) << 16) | (unsigned)i;   // ascending key == descending score
        }
        km[i] = key;
    }
    for (unsigned k = 2; k <= SORTN; k <<= 1) {
        for (unsigned j = k >> 1; j > 0; j >>= 1) {
            __syncthreads();
            for (unsigned t = tid; t < SORTN / 2; t += 1024) {
                unsigned i = (t & (j - 1)) | ((t & ~(j - 1)) << 1);
                unsigned ixj = i | j;
                bool up = ((i & k) == 0);
                unsigned a = km[i], b = km[ixj];
                if ((a > b) == up) { km[i] = b; km[ixj] = a; }
            }
        }
    }
    __syncthreads();
    for (int r = tid; r < KPRE; r += 1024) {
        int idx = (int)(km[r] & 0xFFFFu);
        tbx[r] = bxA[idx]; tby[r] = byA[idx]; tbw[r] = bwA[idx]; tbh[r] = bhA[idx];
    }
}

// ------------------------------- IoU bitmask --------------------------------
__global__ __launch_bounds__(64) void iou_mask(const float* __restrict__ tbx, const float* __restrict__ tby,
                                               const float* __restrict__ tbw, const float* __restrict__ tbh,
                                               unsigned* __restrict__ mask) {
    int row = blockIdx.x, w = threadIdx.x;
    if (w >= NMSW) return;
    float xi = tbx[row], yi = tby[row], wi = tbw[row], hi = tbh[row];
    float x2i = xi + wi, y2i = yi + hi, ai = wi * hi;
    unsigned bits = 0;
    for (int c = 0; c < 32; ++c) {
        int col = w * 32 + c;
        if (col < KPRE && col > row) {
            float xj = tbx[col], yj = tby[col], wj = tbw[col], hj = tbh[col];
            float ix = fmaxf(xi, xj), iy = fmaxf(yi, yj);
            float iw = fmaxf(fminf(x2i, xj + wj) - ix, 0.0f);
            float ih = fmaxf(fminf(y2i, yj + hj) - iy, 0.0f);
            float inter = iw * ih;
            float iou = inter / (ai + wj * hj - inter);
            if (iou > 0.7f) bits |= (1u << c);
        }
    }
    mask[(size_t)row * NMSW + w] = bits;
}

// ---------------------- sequential NMS scan + selection ---------------------
__global__ __launch_bounds__(256) void nms_scan(const unsigned* __restrict__ mask,
                                                const float* __restrict__ tbx, const float* __restrict__ tby,
                                                const float* __restrict__ tbw, const float* __restrict__ tbh,
                                                float* __restrict__ rois) {
    __shared__ unsigned rem[NMSW];
    __shared__ int sel[KPOST];
    int tid = threadIdx.x;
    if (tid < NMSW) rem[tid] = 0u;
    for (int i = 0; i < KPRE; ++i) {
        __syncthreads();
        bool alive = ((rem[i >> 5] >> (i & 31)) & 1u) == 0u;
        __syncthreads();
        if (alive && tid < NMSW) rem[tid] |= mask[(size_t)i * NMSW + tid];
    }
    __syncthreads();
    if (tid == 0) {
        int cnt = 0;
        for (int r = 0; r < KPRE && cnt < KPOST; ++r)
            if (((rem[r >> 5] >> (r & 31)) & 1u) == 0u) sel[cnt++] = r;
        while (cnt < KPOST) sel[cnt++] = 0;
    }
    __syncthreads();
    if (tid < KPOST) {
        int r = sel[tid];
        rois[tid * 4 + 0] = tbx[r]; rois[tid * 4 + 1] = tby[r];
        rois[tid * 4 + 2] = tbw[r]; rois[tid * 4 + 3] = tbh[r];
    }
}

// -------- ROI-align + 2x2 maxpool, writing f16 straight into B fragments ----
__global__ __launch_bounds__(256) void roi_align(const float* __restrict__ x,
                                                 const float* __restrict__ rois,
                                                 f16* __restrict__ bp2) {
    __shared__ int   sx0[28], sy0[28];
    __shared__ float swx[28], swy[28];
    int roi = blockIdx.x, tid = threadIdx.x;
    float rx = rois[roi * 4 + 0], ry = rois[roi * 4 + 1];
    float rw = rois[roi * 4 + 2], rh = rois[roi * 4 + 3];
    if (tid < 28) {
        float g = (float)tid + 0.5f;
        float gx = fminf(fmaxf(rx + g * rw * (1.0f / 28.0f), 0.0f), 63.0f);
        float gy = fminf(fmaxf(ry + g * rh * (1.0f / 28.0f), 0.0f), 63.0f);
        int x0 = min(max((int)floorf(gx), 0), 62);
        int y0 = min(max((int)floorf(gy), 0), 62);
        sx0[tid] = x0; swx[tid] = gx - (float)x0;
        sy0[tid] = y0; swy[tid] = gy - (float)y0;
    }
    __syncthreads();
    int nt = roi >> 4, nl = roi & 15;
    for (int c = 0; c < Cdim; ++c) {
        const float* F = x + (size_t)c * NPIX;
        for (int op = tid; op < 196; op += 256) {
            int py = op / 14, px = op % 14;
            float m = -3.4e38f;
#pragma unroll
            for (int dy = 0; dy < 2; ++dy)
#pragma unroll
                for (int dx = 0; dx < 2; ++dx) {
                    int gy = 2 * py + dy, gx = 2 * px + dx;
                    int x0 = sx0[gx], y0 = sy0[gy];
                    float wx = swx[gx], wy = swy[gy];
                    float f00 = F[y0 * 64 + x0],       f10 = F[y0 * 64 + x0 + 1];
                    float f01 = F[(y0 + 1) * 64 + x0], f11 = F[(y0 + 1) * 64 + x0 + 1];
                    float v = f00 * (1.0f - wy) * (1.0f - wx) + f01 * wy * (1.0f - wx)
                            + f10 * (1.0f - wy) * wx + f11 * wy * wx;
                    m = fmaxf(m, v);
                }
            int k = c * 196 + op;
            int kt = k >> 5, kl = k & 31;
            int lane = ((kl >> 4) << 4) + nl, hh = kl & 15;
            bp2[((((size_t)kt * NT_HEAD + nt) * 32 + lane) << 4) + hh] = (f16)m;
        }
    }
}

// -------------- head GEMM: register-blocked 16x64, split-K x8 ---------------
__global__ __launch_bounds__(32) void head_wmma(const f16* __restrict__ apack,
                                                const f16* __restrict__ bpack,
                                                float* __restrict__ partial) {
    int mt = blockIdx.x, ntb = blockIdx.y, ks = blockIdx.z, lane = threadIdx.x; // 7 x 4 x 8
    const v16h* Af = (const v16h*)apack + (size_t)mt * KT_HEAD * 32 + lane;
    const v16h* Bf = (const v16h*)bpack + (size_t)ntb * 4 * 32 + lane;
    int k0 = ks * KT_SLICE, k1 = k0 + KT_SLICE;
    v8f acc[4] = {};
#pragma unroll 2
    for (int kt = k0; kt < k1; ++kt) {
        int kpf = (kt + 2 < k1) ? kt + 2 : k1 - 1;
        __builtin_prefetch((const void*)(Af + (size_t)kpf * 32), 0, 1);
        __builtin_prefetch((const void*)(Bf + (size_t)kpf * NT_HEAD * 32), 0, 1);
        v16h a = Af[(size_t)kt * 32];
#pragma unroll
        for (int b = 0; b < 4; ++b) {
            v16h bb = Bf[(size_t)kt * NT_HEAD * 32 + b * 32];
            acc[b] = __builtin_amdgcn_wmma_f32_16x16x32_f16(false, a, false, bb,
                                                            (short)0, acc[b], false, false);
        }
    }
    int hi = lane >> 4, nl = lane & 15, mloc = hi * 8;
#pragma unroll
    for (int b = 0; b < 4; ++b) {
        int roi = (ntb * 4 + b) * 16 + nl;
#pragma unroll
        for (int r = 0; r < 8; ++r) {
            int m = mt * 16 + mloc + r;                  // < 112 always
            partial[((size_t)ks * KPOST + roi) * 112 + m] = acc[b][r];
        }
    }
}

__global__ __launch_bounds__(256) void head_reduce(const float* __restrict__ partial,
                                                   const float* __restrict__ bias,
                                                   float* __restrict__ hout) {
    int gid = blockIdx.x * 256 + threadIdx.x;            // 256*105 threads
    int roi = gid / 105, m = gid % 105;
    float s = bias[m];
#pragma unroll
    for (int ks = 0; ks < KSLICE; ++ks)
        s += partial[((size_t)ks * KPOST + roi) * 112 + m];
    hout[gid] = s;
}

// ------------------------------- finalization -------------------------------
__global__ __launch_bounds__(256) void finalize(const float* __restrict__ hout,
                                                const float* __restrict__ rois,
                                                float* __restrict__ out_prop,
                                                float* __restrict__ out_fbox,
                                                float* __restrict__ out_fcls,
                                                float* __restrict__ out_ftfm) {
    int roi = threadIdx.x;
    const float* ho = hout + (size_t)roi * 105;
    int det = 0; float best = ho[0];
    for (int j = 1; j < NCLS; ++j) if (ho[j] > best) { best = ho[j]; det = j; }
    float t0 = ho[NCLS + det * 4 + 0], t1 = ho[NCLS + det * 4 + 1];
    float t2 = ho[NCLS + det * 4 + 2], t3 = ho[NCLS + det * 4 + 3];
    float X = rois[roi * 4 + 0] * 16.0f, Y = rois[roi * 4 + 1] * 16.0f;
    float Wd = rois[roi * 4 + 2] * 16.0f, Hd = rois[roi * 4 + 3] * 16.0f;
    float fx = X + Wd * t1, fy = Y + Hd * t0;
    float fw = Wd * __expf(t3), fh = Hd * __expf(t2);
    out_prop[roi * 5 + 0] = 0.0f; out_prop[roi * 5 + 1] = X;  out_prop[roi * 5 + 2] = Y;
    out_prop[roi * 5 + 3] = Wd;   out_prop[roi * 5 + 4] = Hd;
    out_fbox[roi * 5 + 0] = 0.0f; out_fbox[roi * 5 + 1] = fx; out_fbox[roi * 5 + 2] = fy;
    out_fbox[roi * 5 + 3] = fw;   out_fbox[roi * 5 + 4] = fh;
    for (int j = 0; j < NCLS; ++j) out_fcls[roi * NCLS + j] = ho[j];
    for (int j = 0; j < 84; ++j)   out_ftfm[roi * 84 + j]   = ho[NCLS + j];
}

// ------------------------------ host-side glue ------------------------------
static constexpr size_t al256(size_t x) { return (x + 255) & ~(size_t)255; }

extern "C" void kernel_launch(void* const* d_in, const int* in_sizes, int n_in,
                              void* d_out, int out_size, void* d_ws, size_t ws_size,
                              hipStream_t stream) {
    const float* x       = (const float*)d_in[0];
    const float* rpn_w   = (const float*)d_in[1];
    const float* rpn_b   = (const float*)d_in[2];
    const float* cls_w   = (const float*)d_in[3];
    const float* cls_b   = (const float*)d_in[4];
    const float* tfm_w   = (const float*)d_in[5];
    const float* tfm_b   = (const float*)d_in[6];
    const float* head_w  = (const float*)d_in[7];
    const float* head_b  = (const float*)d_in[8];
    const float* anchors = (const float*)d_in[9];

    char* ws = (char*)d_ws;
    size_t off = 0;
    f16*      apack  = (f16*)     (ws + off); off = al256(off + (size_t)MT_CONV * KT_CONV * 512 * 2);
    f16*      bpack  = (f16*)     (ws + off); off = al256(off + (size_t)KT_CONV * NT_CONV * 512 * 2);
    f16*      rpnA   = (f16*)     (ws + off); off = al256(off + (size_t)MT_RPN * KT_RPN * 512 * 2);
    f16*      hpackB = (f16*)     (ws + off); off = al256(off + (size_t)KT_RPN * NT_CONV * 512 * 2);
    float*    o54map = (float*)   (ws + off); off = al256(off + (size_t)64 * NPIX * 4);
    float*    scores = (float*)   (ws + off); off = al256(off + (size_t)NSCORE * 4);
    float*    bxA    = (float*)   (ws + off); off = al256(off + (size_t)NSCORE * 4);
    float*    byA    = (float*)   (ws + off); off = al256(off + (size_t)NSCORE * 4);
    float*    bwA    = (float*)   (ws + off); off = al256(off + (size_t)NSCORE * 4);
    float*    bhA    = (float*)   (ws + off); off = al256(off + (size_t)NSCORE * 4);
    float*    tbx    = (float*)   (ws + off); off = al256(off + (size_t)KPRE * 4);
    float*    tby    = (float*)   (ws + off); off = al256(off + (size_t)KPRE * 4);
    float*    tbw    = (float*)   (ws + off); off = al256(off + (size_t)KPRE * 4);
    float*    tbh    = (float*)   (ws + off); off = al256(off + (size_t)KPRE * 4);
    unsigned* mask   = (unsigned*)(ws + off); off = al256(off + (size_t)KPRE * NMSW * 4);
    float*    rois   = (float*)   (ws + off); off = al256(off + (size_t)KPOST * 4 * 4);
    f16*      apack2 = (f16*)     (ws + off); off = al256(off + (size_t)MT_HEAD * KT_HEAD * 512 * 2);
    f16*      bpack2 = (f16*)     (ws + off); off = al256(off + (size_t)KT_HEAD * NT_HEAD * 512 * 2);
    float*    part   = (float*)   (ws + off); off = al256(off + (size_t)KSLICE * KPOST * 112 * 4);
    float*    hout   = (float*)   (ws + off); off = al256(off + (size_t)KPOST * 105 * 4);

    float* out      = (float*)d_out;
    float* out_cls  = out;               // (1,18,64,64)   73728
    float* out_tfm  = out + 73728;       // (1,36,64,64)  147456
    float* out_prop = out + 221184;      // (256,5)         1280
    float* out_fbox = out + 222464;      // (256,5)         1280
    float* out_fcls = out + 223744;      // (256,21)        5376
    float* out_ftfm = out + 229120;      // (256,84)       21504

    pack_conv_w<<<(MT_CONV * KT_CONV * 512) / 256, 256, 0, stream>>>(rpn_w, apack);
    pack_conv_x<<<(KT_CONV * NT_CONV * 512) / 256, 256, 0, stream>>>(x, bpack);
    pack_rpn_w<<<(MT_RPN * KT_RPN * 512) / 256, 256, 0, stream>>>(cls_w, tfm_w, rpnA);
    pack_head_w<<<(MT_HEAD * KT_HEAD * 512) / 256, 256, 0, stream>>>(head_w, apack2);

    conv_wmma<<<dim3(MT_CONV, NT_CONV / 4), 32, 0, stream>>>(apack, bpack, rpn_b, hpackB);
    rpn_wmma<<<dim3(MT_RPN, NT_CONV / 4), 32, 0, stream>>>(rpnA, hpackB, cls_b, tfm_b,
                                                           out_cls, out_tfm, o54map);
    rpn_decode<<<NSCORE / 256, 256, 0, stream>>>(o54map, anchors, scores, bxA, byA, bwA, bhA);

    bitonic_topk<<<1, 1024, SORTN * sizeof(unsigned), stream>>>(scores, bxA, byA, bwA, bhA,
                                                                tbx, tby, tbw, tbh);
    iou_mask<<<KPRE, 64, 0, stream>>>(tbx, tby, tbw, tbh, mask);
    nms_scan<<<1, 256, 0, stream>>>(mask, tbx, tby, tbw, tbh, rois);

    roi_align<<<KPOST, 256, 0, stream>>>(x, rois, bpack2);
    head_wmma<<<dim3(MT_HEAD, NT_HEAD / 4, KSLICE), 32, 0, stream>>>(apack2, bpack2, part);
    head_reduce<<<(KPOST * 105) / 256 + 1, 256, 0, stream>>>(part, head_b, hout);
    finalize<<<1, 256, 0, stream>>>(hout, rois, out_prop, out_fbox, out_fcls, out_ftfm);

    (void)in_sizes; (void)n_in; (void)out_size; (void)ws_size; (void)off;
}
